// ClassifierHeadMultiProposal_23587960389890
// MI455X (gfx1250) — compile-verified, matrix operand order checked by reading, not run on hardware
//
#include <hip/hip_runtime.h>
#include <math.h>

// ---------------- problem constants ----------------
#define NN   16
#define AA   5
#define LI   32
#define LQA  64
#define DD   1024
#define BB   (NN * AA)          // 80
#define MM   (NN * AA * LI)     // 2560 rows
#define NEGV (-1e10f)
#define NEGINF (-3.402823466e38f)

typedef __attribute__((ext_vector_type(16))) __bf16 v16bf;
typedef __attribute__((ext_vector_type(8)))  float  v8f;

// ---------------- helpers ----------------
__device__ __forceinline__ unsigned short f2bf(float f) {
    unsigned u = __float_as_uint(f);
    unsigned r = u + 0x7FFFu + ((u >> 16) & 1u);   // round-to-nearest-even
    return (unsigned short)(r >> 16);
}

__device__ __forceinline__ float block_reduce_sum(float v) {
    __shared__ float sred[256];
    int t = threadIdx.x;
    sred[t] = v;
    __syncthreads();
    for (int s = 128; s > 0; s >>= 1) {
        if (t < s) sred[t] += sred[t + s];
        __syncthreads();
    }
    float r = sred[0];
    __syncthreads();
    return r;
}

// ---------------- 1) masked max-pool over Lqa (bandwidth-dominant) ----------------
__global__ __launch_bounds__(256) void pool_kernel(
        const float* __restrict__ statement, const float* __restrict__ smask,
        float* __restrict__ x, float* __restrict__ msk) {
    const int row = blockIdx.x;                 // (n,a,li) flattened, < 2560
    const float* srow = statement + (size_t)row * LQA * DD;
    const float* mrow = smask + (size_t)row * LQA;
    const int d4 = threadIdx.x * 4;
    float bx = NEGINF, by = NEGINF, bz = NEGINF, bw = NEGINF;
    bool anyZero = false, anyOne = false;
    for (int w = 0; w < LQA; ++w) {
        // keep the useful-load stream deep: prefetch 4 slots ahead if unmasked
        if (w + 4 < LQA && mrow[w + 4] != 0.0f)
            __builtin_prefetch(srow + (size_t)(w + 4) * DD + d4, 0, 1);
        float m = mrow[w];                      // uniform across block; broadcast
        if (m != 0.0f) {
            const float4 sv = *(const float4*)(srow + (size_t)w * DD + d4);
            bx = fmaxf(bx, sv.x); by = fmaxf(by, sv.y);
            bz = fmaxf(bz, sv.z); bw = fmaxf(bw, sv.w);
            anyOne = true;
        } else {
            anyZero = true;                     // skip the load entirely
        }
    }
    if (anyZero) {                              // masked entries contribute NEG
        bx = fmaxf(bx, NEGV); by = fmaxf(by, NEGV);
        bz = fmaxf(bz, NEGV); bw = fmaxf(bw, NEGV);
    }
    float4 out = make_float4(bx, by, bz, bw);
    *(float4*)(x + (size_t)row * DD + d4) = out;
    if (threadIdx.x == 0) msk[row] = anyOne ? 1.0f : 0.0f;
}

// ---------------- 2) LayerNorm rows -> bf16 A matrix ----------------
__global__ __launch_bounds__(256) void ln_bf16_kernel(
        const float* __restrict__ x, const float* __restrict__ g,
        const float* __restrict__ b, unsigned short* __restrict__ aln) {
    const int row = blockIdx.x;
    const float* xr = x + (size_t)row * DD;
    float lv[4];
    float s = 0.0f, ss = 0.0f;
    #pragma unroll
    for (int k = 0; k < 4; ++k) {
        int d = threadIdx.x + 256 * k;
        lv[k] = xr[d];
        s += lv[k]; ss += lv[k] * lv[k];
    }
    s  = block_reduce_sum(s);
    ss = block_reduce_sum(ss);
    float mu = s / (float)DD;
    float var = ss / (float)DD - mu * mu;
    float rs = rsqrtf(var + 1e-5f);
    #pragma unroll
    for (int k = 0; k < 4; ++k) {
        int d = threadIdx.x + 256 * k;
        float v = (lv[k] - mu) * rs * g[d] + b[d];
        aln[(size_t)row * DD + d] = f2bf(v);
    }
}

// ---------------- 3) proj_w (K x N) -> Wt bf16 (N x K) ----------------
__global__ __launch_bounds__(256) void wt_kernel(
        const float* __restrict__ w, unsigned short* __restrict__ wt) {
    int i = blockIdx.x * 256 + threadIdx.x;     // < 1M
    int n = i >> 10, k = i & 1023;
    wt[i] = f2bf(w[k * DD + n]);
}

// ---- 4) WMMA bf16 GEMM, 2x2 register-blocked, + bias + ReLU + residual (in place) ----
__device__ __forceinline__ v8f wmma_bf16(const v16bf& a, const v16bf& b, v8f c) {
    return __builtin_amdgcn_wmma_f32_16x16x32_bf16(
            false, a, false, b, (short)0, c, false, false);
}

__global__ __launch_bounds__(256) void gemm_residual_kernel(
        const unsigned* __restrict__ alnU, const unsigned* __restrict__ wtU,
        const float* __restrict__ proj_bias, float* __restrict__ x) {
    const int wave = threadIdx.x >> 5;
    const int lane = threadIdx.x & 31;
    const int tile = blockIdx.x * 8 + wave;     // 2560 super-tiles of 32x32
    const int tmi = tile >> 5;                  // 80 M super-tiles
    const int tni = tile & 31;                  // 32 N super-tiles
    const int half = lane >> 4;
    const int l15 = lane & 15;
    const int m0 = tmi * 32 + l15, m1 = m0 + 16;
    const int n0 = tni * 32 + l15, n1 = n0 + 16;

    v8f c00 = {}, c01 = {}, c10 = {}, c11 = {};
    for (int kk = 0; kk < DD; kk += 32) {
        union { unsigned u[8]; v16bf v; } a0, a1, b0, b1;
        #pragma unroll
        for (int r = 0; r < 8; ++r) {
            // K-pair base for VGPR r: ((r&4)<<2) + 8*half + 2*(r&3)  (ISA 16-bit 16x32 layout)
            int kw = (((r & 4) << 2) + (half << 3) + ((r & 3) << 1) + kk) >> 1;
            a0.u[r] = alnU[(size_t)m0 * (DD / 2) + kw];
            a1.u[r] = alnU[(size_t)m1 * (DD / 2) + kw];
            b0.u[r] = wtU [(size_t)n0 * (DD / 2) + kw];
            b1.u[r] = wtU [(size_t)n1 * (DD / 2) + kw];
        }
        c00 = wmma_bf16(a0.v, b0.v, c00);
        c01 = wmma_bf16(a0.v, b1.v, c01);
        c10 = wmma_bf16(a1.v, b0.v, c10);
        c11 = wmma_bf16(a1.v, b1.v, c11);
    }
    const float pb0 = proj_bias[n0];
    const float pb1 = proj_bias[n1];
    #pragma unroll
    for (int r = 0; r < 8; ++r) {
        // C/D layout: VGPR r -> M = r + 8*half; N = lane&15
        int mg0 = tmi * 32 + r + (half << 3);
        int mg1 = mg0 + 16;
        float v;
        v = c00[r] + pb0; v = v > 0.0f ? v : 0.0f; x[(size_t)mg0 * DD + n0] += v;
        v = c01[r] + pb1; v = v > 0.0f ? v : 0.0f; x[(size_t)mg0 * DD + n1] += v;
        v = c10[r] + pb0; v = v > 0.0f ? v : 0.0f; x[(size_t)mg1 * DD + n0] += v;
        v = c11[r] + pb1; v = v > 0.0f ? v : 0.0f; x[(size_t)mg1 * DD + n1] += v;
    }
}

// ---------------- 5) start/end heads -> temporal (into d_out) ----------------
__global__ __launch_bounds__(256) void head_kernel(
        const float* __restrict__ x,
        const float* __restrict__ st_g, const float* __restrict__ st_b,
        const float* __restrict__ st_w, const float* __restrict__ st_bias,
        const float* __restrict__ ed_g, const float* __restrict__ ed_b,
        const float* __restrict__ ed_w, const float* __restrict__ ed_bias,
        const float* __restrict__ tslm, float* __restrict__ temporal) {
    const int row = blockIdx.x;                 // < 2560
    const float* xr = x + (size_t)row * DD;
    float lv[4];
    float s = 0.0f, ss = 0.0f;
    #pragma unroll
    for (int k = 0; k < 4; ++k) {
        int d = threadIdx.x + 256 * k;
        lv[k] = xr[d];
        s += lv[k]; ss += lv[k] * lv[k];
    }
    s  = block_reduce_sum(s);
    ss = block_reduce_sum(ss);
    float mu = s / (float)DD;
    float rs = rsqrtf(ss / (float)DD - mu * mu + 1e-5f);
    float a_st = 0.0f, a_ed = 0.0f;
    #pragma unroll
    for (int k = 0; k < 4; ++k) {
        int d = threadIdx.x + 256 * k;
        float xn = (lv[k] - mu) * rs;
        a_st += (xn * st_g[d] + st_b[d]) * st_w[d];
        a_ed += (xn * ed_g[d] + ed_b[d]) * ed_w[d];
    }
    a_st = block_reduce_sum(a_st);
    a_ed = block_reduce_sum(a_ed);
    if (threadIdx.x == 0) {
        float t_st = a_st + st_bias[0];
        float t_ed = a_ed + ed_bias[0];
        int n  = row / (AA * LI);
        int li = row & (LI - 1);
        float tm = tslm[n * LI + li];
        temporal[row * 2 + 0] = t_st * tm + (1.0f - tm) * NEGV;
        temporal[row * 2 + 1] = t_ed * tm + (1.0f - tm) * NEGV;
    }
}

// ---------------- 6) softmax + top-1 span per (N,A); also emit targets ----------------
__global__ __launch_bounds__(128) void span_kernel(
        const float* __restrict__ temporal, const int* __restrict__ targets,
        int* __restrict__ spans, float* __restrict__ out) {
    int t = threadIdx.x;
    if (t < NN) out[BB + t] = (float)targets[t];
    if (t >= BB) return;
    float p0[LI], p1[LI];
    float mx0 = NEGINF, mx1 = NEGINF;
    for (int li = 0; li < LI; ++li) {
        p0[li] = temporal[(t * LI + li) * 2 + 0];
        p1[li] = temporal[(t * LI + li) * 2 + 1];
        mx0 = fmaxf(mx0, p0[li]); mx1 = fmaxf(mx1, p1[li]);
    }
    float s0 = 0.0f, s1 = 0.0f;
    for (int li = 0; li < LI; ++li) {
        p0[li] = expf(p0[li] - mx0); s0 += p0[li];
        p1[li] = expf(p1[li] - mx1); s1 += p1[li];
    }
    float i0 = 1.0f / s0, i1 = 1.0f / s1;
    float best = -2.0f; int bi = 0, bj = 0;
    for (int i = 0; i < LI; ++i) {
        float ps = p0[i] * i0;
        for (int j = 0; j < LI; ++j) {
            float v = (j >= i) ? ps * (p1[j] * i1) : -1.0f;
            if (v > best) { best = v; bi = i; bj = j; }   // first occurrence wins
        }
    }
    int st2 = bi - 3; if (st2 < 0) st2 = 0;
    spans[t * 2 + 0] = st2;
    spans[t * 2 + 1] = bj + 4;                  // ed + 1 + EXTRA (exclusive)
}

// ---------------- 7) span/global max pools + cls LN + dot -> answer_scores ----------------
__global__ __launch_bounds__(256) void cls_kernel(
        const float* __restrict__ x, const float* __restrict__ msk,
        const int* __restrict__ spans,
        const float* __restrict__ cls_g, const float* __restrict__ cls_b,
        const float* __restrict__ cls_w, const float* __restrict__ cls_bias,
        float* __restrict__ out) {
    __shared__ float mm[2 * DD];
    const int b = blockIdx.x;                   // < 80
    const int st2 = spans[b * 2 + 0];
    const int ed2 = spans[b * 2 + 1];
    #pragma unroll
    for (int k = 0; k < 4; ++k) {
        int d = threadIdx.x + 256 * k;
        float loc = NEGINF, glob = NEGINF;
        bool zg = false, zl = false;
        for (int li = 0; li < LI; ++li) {
            float m = msk[b * LI + li];
            float xv = x[((size_t)(b * LI + li)) * DD + d];
            if (m != 0.0f) { glob = fmaxf(glob, xv); } else { zg = true; }
            bool span = (li >= st2) && (li < ed2);
            if (m != 0.0f && span) { loc = fmaxf(loc, xv); } else { zl = true; }
        }
        if (zg) glob = fmaxf(glob, NEGV);
        if (zl) loc  = fmaxf(loc,  NEGV);
        mm[d] = loc;          // mm = [loc, glob]
        mm[DD + d] = glob;
    }
    __syncthreads();
    float s = 0.0f, ss = 0.0f;
    #pragma unroll
    for (int k = 0; k < 8; ++k) {
        int d = threadIdx.x + 256 * k;
        float v = mm[d];
        s += v; ss += v * v;
    }
    s  = block_reduce_sum(s);
    ss = block_reduce_sum(ss);
    float mu = s / (float)(2 * DD);
    float rs = rsqrtf(ss / (float)(2 * DD) - mu * mu + 1e-5f);
    float acc = 0.0f;
    #pragma unroll
    for (int k = 0; k < 8; ++k) {
        int d = threadIdx.x + 256 * k;
        acc += ((mm[d] - mu) * rs * cls_g[d] + cls_b[d]) * cls_w[d];
    }
    acc = block_reduce_sum(acc);
    if (threadIdx.x == 0) out[b] = acc + cls_bias[0];
}

// ---------------- launcher ----------------
extern "C" void kernel_launch(void* const* d_in, const int* in_sizes, int n_in,
                              void* d_out, int out_size, void* d_ws, size_t ws_size,
                              hipStream_t stream) {
    (void)in_sizes; (void)n_in; (void)out_size; (void)ws_size;
    const float* statement = (const float*)d_in[0];
    const float* smask     = (const float*)d_in[1];
    const int*   targets   = (const int*)  d_in[2];
    // d_in[3] = ts_labels (unused by the computation)
    const float* tslm      = (const float*)d_in[4];
    const float* proj_g    = (const float*)d_in[5];
    const float* proj_b    = (const float*)d_in[6];
    const float* proj_w    = (const float*)d_in[7];
    const float* proj_bias = (const float*)d_in[8];
    const float* st_g = (const float*)d_in[9];
    const float* st_b = (const float*)d_in[10];
    const float* st_w = (const float*)d_in[11];
    const float* st_bias = (const float*)d_in[12];
    const float* ed_g = (const float*)d_in[13];
    const float* ed_b = (const float*)d_in[14];
    const float* ed_w = (const float*)d_in[15];
    const float* ed_bias = (const float*)d_in[16];
    const float* cls_g = (const float*)d_in[17];
    const float* cls_b = (const float*)d_in[18];
    const float* cls_w = (const float*)d_in[19];
    const float* cls_bias = (const float*)d_in[20];

    float* out = (float*)d_out;                 // [0,80) scores, [80,96) targets, [96,5216) temporal
    float* temporal = out + 96;

    // workspace layout (bytes)
    char* p = (char*)d_ws;
    float*          x    = (float*)p;                                   // 2560*1024*4 = 10,485,760
    unsigned short* aln  = (unsigned short*)(p + 10485760);             // 2560*1024*2 =  5,242,880
    unsigned short* wt   = (unsigned short*)(p + 15728640);             // 1024*1024*2 =  2,097,152
    float*          msk  = (float*)(p + 17825792);                      // 2560*4
    int*            spans= (int*)  (p + 17836032);                      // 80*2*4

    pool_kernel<<<MM, 256, 0, stream>>>(statement, smask, x, msk);
    ln_bf16_kernel<<<MM, 256, 0, stream>>>(x, proj_g, proj_b, aln);
    wt_kernel<<<(DD * DD) / 256, 256, 0, stream>>>(proj_w, wt);
    // 2560 super-tiles (32x32 each), 8 waves per block
    gemm_residual_kernel<<<(MM / 32) * (DD / 32) / 8, 256, 0, stream>>>(
        (const unsigned*)aln, (const unsigned*)wt, proj_bias, x);
    head_kernel<<<MM, 256, 0, stream>>>(x, st_g, st_b, st_w, st_bias,
                                        ed_g, ed_b, ed_w, ed_bias, tslm, temporal);
    span_kernel<<<1, 128, 0, stream>>>(temporal, targets, spans, out);
    cls_kernel<<<BB, 256, 0, stream>>>(x, msk, spans, cls_g, cls_b, cls_w, cls_bias, out);
}
